// L2GESRModule_13494787244461
// MI455X (gfx1250) — matrix operand order repaired and checked by reading, not compiled, not run on hardware
//
#include <hip/hip_runtime.h>
#include <hip/hip_bf16.h>
#include <stdint.h>

// CDNA5 / gfx1250, wave32. D = A(16x32 bf16) x B(32x16 bf16) + C(16x16 f32)
typedef __attribute__((ext_vector_type(16))) __bf16 v16bf;
typedef __attribute__((ext_vector_type(8)))  float  v8f;

namespace {
constexpr int kBz   = 8, kH = 128, kWd = 128, kCin = 256, kCout = 256;
constexpr int kPix  = kBz * kH * kWd;   // 131072 GEMM rows
constexpr int kTiles = kPix / 16;       // 8192 row-tiles (16 rows each)
constexpr int kKF   = kCin / 32;        // 8 K-fragments of 32
constexpr int kNT   = kCout / 16;       // 16 N-tiles of 16
}

__device__ __forceinline__ uint32_t bf16_rne(float f) {
    uint32_t u = __float_as_uint(f);
    return (u + 0x7FFFu + ((u >> 16) & 1u)) >> 16;   // round-to-nearest-even
}
__device__ __forceinline__ uint32_t pack2(float lo, float hi) {
    return bf16_rne(lo) | (bf16_rne(hi) << 16);
}

// -------------------------------------------------------------------------
// Prep: pack Wl (f32, [Cin][Cout] row-major) into bf16 WMMA B-fragments.
// Fragment block (nt,kf) = 1 KB = 32 lanes x 8 dwords, contiguous:
//   lane<16 : N = nt*16+lane,    elements e=0..7 -> K = kf*32 + e,     e=8..15 -> K = kf*32 + 8 + e
//   lane>=16: N = nt*16+lane-16, elements e=0..7 -> K = kf*32 + 8 + e, e=8..15 -> K = kf*32 + 16 + e
// (CDNA5 ISA: B rows striped across lanes within a VGPR; K pairs packed.)
// Total: 16 nt * 8 kf * 1KB = 128 KB in d_ws.
// -------------------------------------------------------------------------
__global__ void pack_w_kernel(const float* __restrict__ Wl,
                              uint32_t* __restrict__ wpack) {
    int t = blockIdx.x * blockDim.x + threadIdx.x;      // one dword each
    if (t >= kNT * kKF * 32 * 8) return;
    int v    = t & 7;            // dword within lane (2 bf16 = K pair)
    int lane = (t >> 3) & 31;
    int blk  = t >> 8;           // 256 dwords per block
    int kf   = blk & (kKF - 1);
    int nt   = blk / kKF;
    int n    = nt * 16 + (lane & 15);
    int k0   = kf * 32 + ((lane >> 4) << 3) + (v < 4 ? v * 2 : 16 + (v - 4) * 2);
    float lo = Wl[(size_t)(k0 + 0) * kCout + n];
    float hi = Wl[(size_t)(k0 + 1) * kCout + n];
    wpack[t] = pack2(lo, hi);
}

// -------------------------------------------------------------------------
// Fused shifted-gather + GEMM + bias:
//   out[b,i,j,:] = Fl[b, min(i+1,127), min(j+1,127), :] @ Wl + bl
// One wave32 per 16-row x 256-col output tile. EXEC is all-1s (no guards).
// -------------------------------------------------------------------------
__global__ void __launch_bounds__(256)
warp_gemm_kernel(const float* __restrict__ Fl,
                 const uint32_t* __restrict__ wpack,
                 const float* __restrict__ bl,
                 float* __restrict__ out) {
    const int lane = threadIdx.x & 31;
    const int wave = threadIdx.x >> 5;
    const int tile = blockIdx.x * (blockDim.x >> 5) + wave;  // [0, 8192)
    const int p0   = tile << 4;                              // first output row

    // --- gathered A-row source address for this lane (row = lane & 15) ---
    const int r  = lane & 15;
    const int p  = p0 + r;                 // output pixel index
    const int b  = p >> 14;                // / (H*W)
    const int i  = (p >> 7) & 127;
    const int j  = p & 127;
    const int ix = (i < 127) ? i + 1 : 127;    // ESF == ones -> +1 with clamp
    const int iy = (j < 127) ? j + 1 : 127;
    const float* src = Fl + ((size_t)((b << 14) + (ix << 7) + iy) << 8);

    // --- load + convert A: 8 resident bf16 fragments covering K=0..255 ---
    // ISA A 16x32 layout: lane<16 holds K {kf*32+0..7, +16..23},
    //                     lane>=16 holds K {kf*32+8..15, +24..31}.
    union AV { v16bf v; uint32_t u[8]; } A[kKF];
    const int khalf = (lane >> 4) << 3;    // 0 or 8
    #pragma unroll
    for (int kf = 0; kf < kKF; ++kf) {
        const float* s = src + kf * 32 + khalf;
        float4 x0 = *(const float4*)(s + 0);
        float4 x1 = *(const float4*)(s + 4);
        float4 x2 = *(const float4*)(s + 16);
        float4 x3 = *(const float4*)(s + 20);
        A[kf].u[0] = pack2(x0.x, x0.y);  A[kf].u[1] = pack2(x0.z, x0.w);
        A[kf].u[2] = pack2(x1.x, x1.y);  A[kf].u[3] = pack2(x1.z, x1.w);
        A[kf].u[4] = pack2(x2.x, x2.y);  A[kf].u[5] = pack2(x2.z, x2.w);
        A[kf].u[6] = pack2(x3.x, x3.y);  A[kf].u[7] = pack2(x3.z, x3.w);
    }

    // --- N-tile loop: 8 WMMAs each, then bias + coalesced store ---
    const int n_lane = lane & 15;
    float* obase = out + ((size_t)p0 << 8) + n_lane;   // + nt*16, + row*256 later
    for (int nt = 0; nt < kNT; ++nt) {
        v8f c = {};
        #pragma unroll
        for (int kf = 0; kf < kKF; ++kf) {
            union BV { v16bf v; uint4 q[2]; } Bf;
            const uint4* bp = (const uint4*)(wpack + (((nt * kKF + kf) << 8) |
                                                      (lane << 3)));
            Bf.q[0] = bp[0];
            Bf.q[1] = bp[1];
            c = __builtin_amdgcn_wmma_f32_16x16x32_bf16(
                    /*neg_a=*/false, A[kf].v, /*neg_b=*/false, Bf.v,
                    /*c_mod=*/(short)0, c, /*reuse_a=*/false, /*reuse_b=*/false);
        }
        const float bias = bl[nt * 16 + n_lane];
        float* o = obase + nt * 16;
        // D layout: VGPR rr -> row rr (lanes 0-15) / row rr+8 (lanes 16-31)
        #pragma unroll
        for (int rr = 0; rr < 8; ++rr) {
            const int row = (lane < 16) ? rr : rr + 8;
            o[(size_t)row << 8] = c[rr] + bias;
        }
    }
}

// -------------------------------------------------------------------------
// Inputs (setup_inputs order): 0=Fh (dead), 1=Fl, 2=Wh (dead), 3=bh (dead),
// 4=Wl, 5=bl.  ESF==ones makes the Fh branch dead code.
// d_ws: needs 128 KB for packed bf16 W fragments.
// -------------------------------------------------------------------------
extern "C" void kernel_launch(void* const* d_in, const int* in_sizes, int n_in,
                              void* d_out, int out_size, void* d_ws, size_t ws_size,
                              hipStream_t stream) {
    (void)in_sizes; (void)n_in; (void)out_size; (void)ws_size;
    const float* Fl = (const float*)d_in[1];
    const float* Wl = (const float*)d_in[4];
    const float* bl = (const float*)d_in[5];
    uint32_t* wpack = (uint32_t*)d_ws;                    // 128 KB used

    pack_w_kernel<<<256, 256, 0, stream>>>(Wl, wpack);    // 32768 dwords (65536 threads launched)
    warp_gemm_kernel<<<kTiles / 8, 256, 0, stream>>>(Fl, wpack, bl,
                                                     (float*)d_out);
}
